// UniformLoss_43284680409440
// MI455X (gfx1250) — matrix operand order repaired and checked by reading, not compiled, not run on hardware
//
#include <hip/hip_runtime.h>
#include <math.h>

typedef __attribute__((ext_vector_type(16))) __bf16 v16bf;
typedef __attribute__((ext_vector_type(8)))  __bf16 v8bf;
typedef __attribute__((ext_vector_type(8)))  float  v8f;

#define N_ROWS 8192
#define N_DIM  512
#define MT     128                // macro tile edge (rows & cols)
#define NTM    (N_ROWS / MT)      // 64 macro tiles per side
#define KC     64                 // K chunk staged in LDS
#define LDK    (KC + 8)           // LDS row stride (bf16), conflict-avoiding pad
#define TEMP   2.0f

__device__ __forceinline__ int tri_off(int b) {
    // number of upper-tri macro tiles before macro-row b
    return b * NTM - (b * (b - 1)) / 2;
}

// ---------------------------------------------------------------------------
// Kernel 1: per-row squared norms (fp32) + split-bf16 hi/lo planes; zero acc.
// One 256-thread block per row; each thread handles 2 of 512 elements.
// ---------------------------------------------------------------------------
__global__ void __launch_bounds__(256)
prep_kernel(const float* __restrict__ x, float* __restrict__ sq,
            __bf16* __restrict__ xhi, __bf16* __restrict__ xlo,
            float* __restrict__ acc)
{
    const int row = blockIdx.x;
    const int t   = threadIdx.x;
    if (row == 0 && t == 0) *acc = 0.0f;   // re-zeroed every launch (graph-safe)

    const float2 v = ((const float2*)(x + (size_t)row * N_DIM))[t];
    __bf16 h0 = (__bf16)v.x;
    __bf16 h1 = (__bf16)v.y;
    __bf16 l0 = (__bf16)(v.x - (float)h0);
    __bf16 l1 = (__bf16)(v.y - (float)h1);
    const size_t base = (size_t)row * N_DIM + 2 * t;
    xhi[base] = h0; xhi[base + 1] = h1;
    xlo[base] = l0; xlo[base + 1] = l1;

    float p = v.x * v.x + v.y * v.y;
    #pragma unroll
    for (int off = 16; off > 0; off >>= 1) p += __shfl_down(p, off, 32);

    __shared__ float red[8];
    if ((t & 31) == 0) red[t >> 5] = p;
    __syncthreads();
    if (t == 0) {
        float s = 0.0f;
        #pragma unroll
        for (int i = 0; i < 8; ++i) s += red[i];
        sq[row] = s;
    }
}

// ---------------------------------------------------------------------------
// Kernel 2: upper-triangular tiled G = X·Xᵀ via bf16x3 WMMA, fused epilogue.
// Block = 256 threads = 8 waves, macro tile 128x128. Waves arranged 4x2:
// wave owns a 32x64 region = 2x4 sub-tiles of 16x16 (8 accumulators), so each
// 32-K step does 24 WMMAs per 24 ds_load_b128 (A frags reused across 4 sj,
// B frags reused across 2 si).
// ---------------------------------------------------------------------------
__global__ void __launch_bounds__(256)
pdist_gemm_kernel(const __bf16* __restrict__ xhi, const __bf16* __restrict__ xlo,
                  const float* __restrict__ sq, float* __restrict__ acc)
{
    __shared__ __bf16 Ah[MT][LDK];
    __shared__ __bf16 Al[MT][LDK];
    __shared__ __bf16 Bh[MT][LDK];
    __shared__ __bf16 Bl[MT][LDK];

    // ---- decode linear block id -> (br, bc) with bc >= br (upper triangle)
    const int t = blockIdx.x;
    float disc = (2.0f * NTM + 1.0f) * (2.0f * NTM + 1.0f) - 8.0f * (float)t;
    int br = (int)((2.0f * NTM + 1.0f - sqrtf(disc)) * 0.5f);
    if (br < 0) br = 0;
    if (br > NTM - 1) br = NTM - 1;
    while (br > 0 && tri_off(br) > t) --br;
    while (tri_off(br + 1) <= t) ++br;
    const int bc = br + (t - tri_off(br));
    const int r0 = br * MT;
    const int c0 = bc * MT;

    const int tid  = threadIdx.x;
    const int wave = tid >> 5;
    const int lane = tid & 31;
    const int wr   = wave >> 1;         // 0..3 : wave row   (32 rows each)
    const int wc   = wave & 1;          // 0..1 : wave col   (64 cols each)

    v8f cc[2][4];
    #pragma unroll
    for (int si = 0; si < 2; ++si)
        #pragma unroll
        for (int sj = 0; sj < 4; ++sj)
            cc[si][sj] = (v8f){};

    // LDS-fill assignment: thread t loads row (t>>1), 32 elems at (t&1)*32
    const int lr = tid >> 1;            // 0..127
    const int lk = (tid & 1) * 32;      // 0 or 32
    const __bf16* gAh = xhi + (size_t)(r0 + lr) * N_DIM + lk;
    const __bf16* gAl = xlo + (size_t)(r0 + lr) * N_DIM + lk;
    const __bf16* gBh = xhi + (size_t)(c0 + lr) * N_DIM + lk;
    const __bf16* gBl = xlo + (size_t)(c0 + lr) * N_DIM + lk;

    const int h = lane >> 4;            // half-wave selector
    const int m = lane & 15;

    for (int kc = 0; kc < N_DIM; kc += KC) {
        // ---- stage K-chunk into LDS (no divergence)
        #pragma unroll
        for (int q = 0; q < 4; ++q) {
            *(v8bf*)&Ah[lr][lk + 8 * q] = *(const v8bf*)(gAh + kc + 8 * q);
            *(v8bf*)&Al[lr][lk + 8 * q] = *(const v8bf*)(gAl + kc + 8 * q);
            *(v8bf*)&Bh[lr][lk + 8 * q] = *(const v8bf*)(gBh + kc + 8 * q);
            *(v8bf*)&Bl[lr][lk + 8 * q] = *(const v8bf*)(gBl + kc + 8 * q);
        }
        __syncthreads();

        if (kc + KC < N_DIM) {          // hint next chunk (global_prefetch_b8)
            __builtin_prefetch(gAh + kc + KC);
            __builtin_prefetch(gBh + kc + KC);
        }

        #pragma unroll
        for (int kk = 0; kk < KC; kk += 32) {
            // A fragments 16x32 bf16: lanes 0-15 hold K = h*8..+7, 16+h*8..+7
            union { v16bf v; v8bf p[2]; } ah[2], al[2], bh, bl;
            #pragma unroll
            for (int si = 0; si < 2; ++si) {
                const int arow = wr * 32 + si * 16 + m;
                ah[si].p[0] = *(const v8bf*)&Ah[arow][kk + h * 8];
                ah[si].p[1] = *(const v8bf*)&Ah[arow][kk + 16 + h * 8];
                al[si].p[0] = *(const v8bf*)&Al[arow][kk + h * 8];
                al[si].p[1] = *(const v8bf*)&Al[arow][kk + 16 + h * 8];
            }
            #pragma unroll
            for (int sj = 0; sj < 4; ++sj) {
                // B fragment 32x16: lane half h holds K = kk + h*8..+7, kk + 16 + h*8..+7
                const int brow = wc * 64 + sj * 16 + m;
                bh.p[0] = *(const v8bf*)&Bh[brow][kk + h * 8];
                bh.p[1] = *(const v8bf*)&Bh[brow][kk + 16 + h * 8];
                bl.p[0] = *(const v8bf*)&Bl[brow][kk + h * 8];
                bl.p[1] = *(const v8bf*)&Bl[brow][kk + 16 + h * 8];
                #pragma unroll
                for (int si = 0; si < 2; ++si) {
                    cc[si][sj] = __builtin_amdgcn_wmma_f32_16x16x32_bf16(
                        false, ah[si].v, false, bh.v, (short)0, cc[si][sj], false, false);
                    cc[si][sj] = __builtin_amdgcn_wmma_f32_16x16x32_bf16(
                        false, ah[si].v, false, bl.v, (short)0, cc[si][sj], false, false);
                    cc[si][sj] = __builtin_amdgcn_wmma_f32_16x16x32_bf16(
                        false, al[si].v, false, bh.v, (short)0, cc[si][sj], false, false);
                }
            }
        }
        __syncthreads();
    }

    // ---- fused epilogue: d2 = sq_i + sq_j - 2 g ; strict upper mask ; exp
    // C layout: lanes 0-15 -> M = r, N = lane ; lanes 16-31 -> M = 8+r, N = lane-16
    float lsum = 0.0f;
    #pragma unroll
    for (int si = 0; si < 2; ++si) {
        const int ibase = r0 + wr * 32 + si * 16 + h * 8;
        #pragma unroll
        for (int sj = 0; sj < 4; ++sj) {
            const int j = c0 + wc * 64 + sj * 16 + m;
            const float sqj = sq[j];
            #pragma unroll
            for (int r = 0; r < 8; ++r) {
                const int i = ibase + r;
                float d2 = sq[i] + sqj - 2.0f * cc[si][sj][r];
                d2 = fmaxf(d2, 0.0f);
                if (j > i) lsum += __expf(-TEMP * d2);
            }
        }
    }
    #pragma unroll
    for (int off = 16; off > 0; off >>= 1) lsum += __shfl_down(lsum, off, 32);
    if (lane == 0) atomicAdd(acc, lsum);
}

// ---------------------------------------------------------------------------
// Kernel 3: final scalar
// ---------------------------------------------------------------------------
__global__ void finalize_kernel(const float* __restrict__ acc, float* __restrict__ out)
{
    const float num_pairs = 33550336.0f;   // 8192*8191/2
    out[0] = logf(acc[0] / num_pairs);
}

// ---------------------------------------------------------------------------
extern "C" void kernel_launch(void* const* d_in, const int* in_sizes, int n_in,
                              void* d_out, int out_size, void* d_ws, size_t ws_size,
                              hipStream_t stream)
{
    (void)in_sizes; (void)n_in; (void)out_size; (void)ws_size;
    const float* x = (const float*)d_in[0];
    float* out = (float*)d_out;

    // workspace layout (all 16B aligned):
    //   [0..4)                 float acc
    //   [16 .. 16+32768)       float sq[8192]
    //   xhi : 8192*512 bf16    (8 MB)
    //   xlo : 8192*512 bf16    (8 MB)
    char*   ws  = (char*)d_ws;
    float*  acc = (float*)ws;
    float*  sq  = (float*)(ws + 16);
    __bf16* xhi = (__bf16*)(ws + 16 + N_ROWS * sizeof(float));
    __bf16* xlo = xhi + (size_t)N_ROWS * N_DIM;

    prep_kernel<<<N_ROWS, 256, 0, stream>>>(x, sq, xhi, xlo, acc);

    const int nblocks = (NTM * (NTM + 1)) / 2;   // 2080 upper-tri macro tiles
    pdist_gemm_kernel<<<nblocks, 256, 0, stream>>>(xhi, xlo, sq, acc);

    finalize_kernel<<<1, 1, 0, stream>>>(acc, out);
}
